// DNC_23304492548263
// MI455X (gfx1250) — matrix operand order repaired and checked by reading, not compiled, not run on hardware
//
#include <hip/hip_runtime.h>
#include <hip/hip_bf16.h>
#include <math.h>

// ---------------- dims ----------------
#define Bb 64
#define Vv 64
#define Hh 512
#define Oo 64
#define Nn 1024
#define Wd 64
#define Rr 4
#define XIx 471
#define EPSf 1e-6f

typedef __attribute__((ext_vector_type(16))) __bf16 v16bf;
typedef __attribute__((ext_vector_type(8)))  float  v8f;
typedef __attribute__((ext_vector_type(4)))  float  f4;

__device__ __forceinline__ float sigf(float x){ return 1.f/(1.f+expf(-x)); }
__device__ __forceinline__ float softplusf(float x){ return (x>20.f)? x : log1pf(expf(x)); }

// ---------------- WMMA GEMM 1: LSTM gates ----------------
// gates[64,2048] = concat(x, read_flat)[64,320] @ W_ih^T + h0[64,512] @ W_hh^T
__device__ __forceinline__ float inp_at(const float* x, const float* rv, int b, int k){
  if (k < Vv) return x[b*Vv + k];
  int kk = k - Vv;                  // read_flat[b, r*Wd + w] = rv[b, w, r]
  return rv[b*(Rr*Wd) + (kk & 63)*Rr + (kk >> 6)];
}

__global__ void k_gates(const float* __restrict__ x, const float* __restrict__ rv,
                        const float* __restrict__ h0,
                        const float* __restrict__ Wih, const float* __restrict__ Whh,
                        float* __restrict__ gates){
  const int j0 = blockIdx.x * 16;    // gate column tile
  const int b0 = blockIdx.y * 16;    // batch row tile
  const int l  = threadIdx.x;        // wave32 lane
  const int arow = b0 + (l & 15);
  const int akhi = (l >> 4) * 8;     // A: K sub-block per lane group
  const int bcol = j0 + (l & 15);
  const int bko  = (l >> 4) * 8;     // B: K offset per lane group
  v8f acc = {};
  // phase 1: input half, Ktot=320, B[k,n]=W_ih[n,k]
  const float* wrow1 = Wih + (size_t)bcol * 320;
  for (int k0 = 0; k0 < 320; k0 += 32){
    v16bf a, bm;
    #pragma unroll
    for (int e = 0; e < 8; ++e){
      a[e]     = (__bf16)inp_at(x, rv, arow, k0 + akhi + e);
      a[8 + e] = (__bf16)inp_at(x, rv, arow, k0 + 16 + akhi + e);
    }
    #pragma unroll
    for (int e = 0; e < 16; ++e) bm[e] = (__bf16)wrow1[k0 + bko + e + (e >= 8 ? 8 : 0)];
    acc = __builtin_amdgcn_wmma_f32_16x16x32_bf16(false, a, false, bm, (short)0, acc, false, false);
  }
  // phase 2: hidden half, Ktot=512, B[k,n]=W_hh[n,k]
  const float* wrow2 = Whh + (size_t)bcol * 512;
  const float* hrow  = h0 + (size_t)arow * 512;
  for (int k0 = 0; k0 < 512; k0 += 32){
    v16bf a, bm;
    #pragma unroll
    for (int e = 0; e < 8; ++e){
      a[e]     = (__bf16)hrow[k0 + akhi + e];
      a[8 + e] = (__bf16)hrow[k0 + 16 + akhi + e];
    }
    #pragma unroll
    for (int e = 0; e < 16; ++e) bm[e] = (__bf16)wrow2[k0 + bko + e + (e >= 8 ? 8 : 0)];
    acc = __builtin_amdgcn_wmma_f32_16x16x32_bf16(false, a, false, bm, (short)0, acc, false, false);
  }
  const int ccol  = j0 + (l & 15);
  const int rbase = b0 + (l >> 4) * 8;
  #pragma unroll
  for (int i = 0; i < 8; ++i) gates[(size_t)(rbase + i) * 2048 + ccol] = acc[i];
}

// ---------------- LSTM activations ----------------
__global__ void k_lstm(const float* __restrict__ gates, const float* __restrict__ blstm,
                       const float* __restrict__ c0, float* __restrict__ h, float* __restrict__ c){
  int idx = blockIdx.x * blockDim.x + threadIdx.x;
  if (idx >= Bb * Hh) return;
  int b = idx >> 9, j = idx & 511;
  const float* g = gates + (size_t)b * 2048;
  float gi = g[j]        + blstm[j];
  float gf = g[512 + j]  + blstm[512 + j];
  float gg = g[1024 + j] + blstm[1024 + j];
  float go = g[1536 + j] + blstm[1536 + j];
  float cv = sigf(gf) * c0[idx] + sigf(gi) * tanhf(gg);
  float hv = sigf(go) * tanhf(cv);
  c[idx] = cv; h[idx] = hv;
}

// ---------------- WMMA GEMM 2: vu = h@W_v, xi = h@W_xi (fused, 535 cols) ----------------
__global__ void k_hproj(const float* __restrict__ h, const float* __restrict__ Wv,
                        const float* __restrict__ Wxi, float* __restrict__ vu, float* __restrict__ xi){
  const int j0 = blockIdx.x * 16; const int b0 = blockIdx.y * 16;
  const int l = threadIdx.x;
  const int arow = b0 + (l & 15);
  const int akhi = (l >> 4) * 8;
  const int bcol = j0 + (l & 15);
  const int bko  = (l >> 4) * 8;
  v8f acc = {};
  const float* hrow = h + (size_t)arow * 512;
  for (int k0 = 0; k0 < 512; k0 += 32){
    v16bf a, bm;
    #pragma unroll
    for (int e = 0; e < 8; ++e){
      a[e]     = (__bf16)hrow[k0 + akhi + e];
      a[8 + e] = (__bf16)hrow[k0 + 16 + akhi + e];
    }
    #pragma unroll
    for (int e = 0; e < 16; ++e){
      int k = k0 + bko + e + (e >= 8 ? 8 : 0);
      float w = 0.f;
      if (bcol < Oo)              w = Wv[(size_t)k * Oo + bcol];
      else if (bcol < Oo + XIx)   w = Wxi[(size_t)k * XIx + (bcol - Oo)];
      bm[e] = (__bf16)w;
    }
    acc = __builtin_amdgcn_wmma_f32_16x16x32_bf16(false, a, false, bm, (short)0, acc, false, false);
  }
  const int ccol = j0 + (l & 15); const int rbase = b0 + (l >> 4) * 8;
  #pragma unroll
  for (int i = 0; i < 8; ++i){
    int row = rbase + i;
    if (ccol < Oo)            vu[row * Oo + ccol] = acc[i];
    else if (ccol < Oo + XIx) xi[row * XIx + (ccol - Oo)] = acc[i];
  }
}

// ---------------- parse xi interface vector ----------------
__global__ void k_parse(const float* __restrict__ xi, float* rstr, float* wstr, float* erase_o,
                        float* freeg, float* ag, float* wg, float* modes, float* wvec){
  int b = blockIdx.x; int t = threadIdx.x;  // 64 threads
  const float* xb = xi + (size_t)b * XIx;
  erase_o[b * Wd + t] = sigf(xb[325 + t]);
  wvec[b * Wd + t]    = xb[389 + t];         // aligned copy of write vector
  if (t < Rr)  rstr[b * Rr + t] = 1.f + softplusf(xb[256 + t]);
  if (t == 0)  wstr[b] = 1.f + softplusf(xb[324]);
  if (t < Rr)  freeg[b * Rr + t] = sigf(xb[453 + t]);
  if (t == 0)  ag[b] = sigf(xb[457]);
  if (t == 0)  wg[b] = sigf(xb[458]);
  if (t < Rr){
    float m0 = xb[459 + t], m1 = xb[463 + t], m2 = xb[467 + t];
    float mx = fmaxf(m0, fmaxf(m1, m2));
    float e0 = expf(m0 - mx), e1 = expf(m1 - mx), e2 = expf(m2 - mx);
    float inv = 1.f / (e0 + e1 + e2);
    modes[b * 12 + t] = e0 * inv; modes[b * 12 + 4 + t] = e1 * inv; modes[b * 12 + 8 + t] = e2 * inv;
  }
}

// ---------------- usage ----------------
__global__ void k_usage(const float* __restrict__ freeg, const float* __restrict__ wr,
                        const float* __restrict__ usage, const float* __restrict__ ww,
                        float* __restrict__ u){
  int idx = blockIdx.x * blockDim.x + threadIdx.x;
  if (idx >= Bb * Nn) return;
  int b = idx >> 10;
  f4 w = ((const f4*)wr)[idx];
  float psi = 1.f;
  #pragma unroll
  for (int r = 0; r < Rr; ++r) psi *= (1.f - freeg[b * Rr + r] * w[r]);
  float us = usage[idx], wv = ww[idx];
  u[idx] = (us + wv - us * wv) * psi;
}

// ---------------- sort + allocation weighting (per batch bitonic) ----------------
__global__ void k_sort_alloc(const float* __restrict__ u, float* __restrict__ a){
  __shared__ float v[Nn];
  __shared__ unsigned short id[Nn];
  int b = blockIdx.x; int t = threadIdx.x;   // 512 threads
  for (int i = t; i < Nn; i += 512){ v[i] = u[b * Nn + i]; id[i] = (unsigned short)i; }
  __syncthreads();
  for (int k = 2; k <= Nn; k <<= 1){
    for (int j = k >> 1; j > 0; j >>= 1){
      for (int i = t; i < Nn; i += 512){
        int ixj = i ^ j;
        if (ixj > i){
          bool up = ((i & k) == 0);
          float vi = v[i], vx = v[ixj];
          if (up ? (vi > vx) : (vi < vx)){
            v[i] = vx; v[ixj] = vi;
            unsigned short ti = id[i]; id[i] = id[ixj]; id[ixj] = ti;
          }
        }
      }
      __syncthreads();
    }
  }
  if (t == 0){
    float run = 1.f;
    for (int i = 0; i < Nn; ++i){ float us = v[i]; a[b * Nn + id[i]] = (1.f - us) * run; run *= us; }
  }
}

// ---------------- write content addressing, ww_n, prec_n ----------------
__global__ void k_write_addr(const float* __restrict__ M, const float* __restrict__ xi,
                             const float* __restrict__ alloc, const float* __restrict__ prec,
                             const float* __restrict__ wstr_p, const float* __restrict__ wg_p,
                             const float* __restrict__ ag_p,
                             float* __restrict__ wwn_out, float* __restrict__ precn_out){
  int b = blockIdx.x; int t = threadIdx.x;  // 256 threads
  __shared__ float s[Nn];
  __shared__ float red[256];
  __shared__ float wk[Wd];
  if (t < Wd) wk[t] = xi[(size_t)b * XIx + 260 + t];
  float p = 0.f;
  if (t < Wd){ float w = xi[(size_t)b * XIx + 260 + t]; p = w * w; }
  red[t] = p; __syncthreads();
  for (int o = 128; o > 0; o >>= 1){ if (t < o) red[t] += red[t + o]; __syncthreads(); }
  float kwn = sqrtf(red[0]);
  __syncthreads();
  float wstr = wstr_p[b];
  for (int i = 0; i < 4; ++i){
    int n = t + 256 * i;
    const f4* Mr = (const f4*)(M + ((size_t)b * Nn + n) * Wd);
    float nn = 0.f, d = 0.f;
    #pragma unroll 4
    for (int w4 = 0; w4 < 16; ++w4){
      f4 m = Mr[w4];
      #pragma unroll
      for (int e = 0; e < 4; ++e){ nn += m[e] * m[e]; d += m[e] * wk[w4 * 4 + e]; }
    }
    s[n] = wstr * d / (sqrtf(nn) * kwn + EPSf);
  }
  __syncthreads();
  float mx = -1e30f;
  for (int i = 0; i < 4; ++i) mx = fmaxf(mx, s[t + 256 * i]);
  red[t] = mx; __syncthreads();
  for (int o = 128; o > 0; o >>= 1){ if (t < o) red[t] = fmaxf(red[t], red[t + o]); __syncthreads(); }
  mx = red[0]; __syncthreads();
  float ls = 0.f;
  for (int i = 0; i < 4; ++i){ int n = t + 256 * i; float e = expf(s[n] - mx); s[n] = e; ls += e; }
  red[t] = ls; __syncthreads();
  for (int o = 128; o > 0; o >>= 1){ if (t < o) red[t] += red[t + o]; __syncthreads(); }
  float inv = 1.f / red[0];
  __syncthreads();
  float ag = ag_p[b], wg = wg_p[b];
  float part = 0.f;
  for (int i = 0; i < 4; ++i){
    int n = t + 256 * i;
    float cw = s[n] * inv;
    float w  = wg * (ag * alloc[b * Nn + n] + (1.f - ag) * cw);
    s[n] = w;
    wwn_out[b * Nn + n] = w;
    part += w;
  }
  red[t] = part; __syncthreads();
  for (int o = 128; o > 0; o >>= 1){ if (t < o) red[t] += red[t + o]; __syncthreads(); }
  float wwsum = red[0];
  for (int i = 0; i < 4; ++i){
    int n = t + 256 * i;
    precn_out[b * Nn + n] = (1.f - wwsum) * prec[b * Nn + n] + s[n];
  }
}

// ---------------- memory write (vectorized elementwise) ----------------
__global__ void k_mwrite(const float* __restrict__ M, const float* __restrict__ wwn,
                         const float* __restrict__ erase, const float* __restrict__ wvec,
                         float* __restrict__ Mn){
  int idx = blockIdx.x * blockDim.x + threadIdx.x;  // B*N*Wd/4
  if (idx >= Bb * Nn * (Wd / 4)) return;
  int w4 = idx & 15; size_t bn = (size_t)idx >> 4; int b = (int)(bn >> 10);
  f4 m  = ((const f4*)M)[idx];
  f4 er = ((const f4*)erase)[b * 16 + w4];
  f4 wv = ((const f4*)wvec)[b * 16 + w4];
  float ww = wwn[bn];
  ((f4*)Mn)[idx] = m * (1.f - ww * er) + ww * wv;
}

// ---------------- read content addressing (softmax per r over N) ----------------
__global__ void k_read_addr(const float* __restrict__ Mn, const float* __restrict__ xi,
                            const float* __restrict__ rstr_p, float* __restrict__ cr){
  int b = blockIdx.x; int t = threadIdx.x;  // 256 threads
  __shared__ float s[Nn * Rr];    // 16 KB
  __shared__ float red4[256 * 4]; // 4 KB
  __shared__ float rk[Wd * Rr];   // 1 KB : read keys staged
  if (t < Wd * Rr) rk[t] = xi[(size_t)b * XIx + t];
  float acc0[4] = {0, 0, 0, 0};
  if (t < Wd){
    for (int r = 0; r < Rr; ++r){ float v = xi[(size_t)b * XIx + t * Rr + r]; acc0[r] = v * v; }
  }
  for (int r = 0; r < Rr; ++r) red4[t * 4 + r] = acc0[r];
  __syncthreads();
  for (int o = 128; o > 0; o >>= 1){
    if (t < o) for (int r = 0; r < Rr; ++r) red4[t * 4 + r] += red4[(t + o) * 4 + r];
    __syncthreads();
  }
  float rkn[4]; for (int r = 0; r < Rr; ++r) rkn[r] = sqrtf(red4[r]);
  __syncthreads();
  float rs[4]; for (int r = 0; r < Rr; ++r) rs[r] = rstr_p[b * Rr + r];
  for (int i = 0; i < 4; ++i){
    int n = t + 256 * i;
    const f4* Mr = (const f4*)(Mn + ((size_t)b * Nn + n) * Wd);
    float nn = 0.f, d[4] = {0, 0, 0, 0};
    #pragma unroll 4
    for (int w4 = 0; w4 < 16; ++w4){
      f4 m = Mr[w4];
      #pragma unroll
      for (int e = 0; e < 4; ++e){
        float mv = m[e]; nn += mv * mv;
        int w = w4 * 4 + e;
        #pragma unroll
        for (int r = 0; r < Rr; ++r) d[r] += mv * rk[w * Rr + r];
      }
    }
    float nrm = sqrtf(nn);
    for (int r = 0; r < Rr; ++r) s[n * Rr + r] = rs[r] * d[r] / (nrm * rkn[r] + EPSf);
  }
  __syncthreads();
  float mx[4] = {-1e30f, -1e30f, -1e30f, -1e30f};
  for (int i = 0; i < 4; ++i){ int n = t + 256 * i; for (int r = 0; r < Rr; ++r) mx[r] = fmaxf(mx[r], s[n * Rr + r]); }
  for (int r = 0; r < Rr; ++r) red4[t * 4 + r] = mx[r];
  __syncthreads();
  for (int o = 128; o > 0; o >>= 1){
    if (t < o) for (int r = 0; r < Rr; ++r) red4[t * 4 + r] = fmaxf(red4[t * 4 + r], red4[(t + o) * 4 + r]);
    __syncthreads();
  }
  for (int r = 0; r < Rr; ++r) mx[r] = red4[r];
  __syncthreads();
  float sm[4] = {0, 0, 0, 0};
  for (int i = 0; i < 4; ++i){
    int n = t + 256 * i;
    for (int r = 0; r < Rr; ++r){ float e = expf(s[n * Rr + r] - mx[r]); s[n * Rr + r] = e; sm[r] += e; }
  }
  for (int r = 0; r < Rr; ++r) red4[t * 4 + r] = sm[r];
  __syncthreads();
  for (int o = 128; o > 0; o >>= 1){
    if (t < o) for (int r = 0; r < Rr; ++r) red4[t * 4 + r] += red4[(t + o) * 4 + r];
    __syncthreads();
  }
  float inv[4]; for (int r = 0; r < Rr; ++r) inv[r] = 1.f / red4[r];
  for (int i = 0; i < 4; ++i){
    int n = t + 256 * i;
    for (int r = 0; r < Rr; ++r) cr[((size_t)b * Nn + n) * Rr + r] = s[n * Rr + r] * inv[r];
  }
}

// ---------------- fused link update + fwd + bwd partials ----------------
// One pass over `link` (268 MB): b128 NT loads, b128 NT stores of L, bwd kept
// in registers (column ownership fixed per thread), fwd via wave32 shfl_xor
// butterfly + one cross-wave combine after the row loop.
#define TM 64
__global__ void k_link(const float* __restrict__ link, const float* __restrict__ prec,
                       const float* __restrict__ wwn, const float* __restrict__ wr,
                       float* __restrict__ Lout, float* __restrict__ fwd, float* __restrict__ bwdpart){
  int b = blockIdx.y, p = blockIdx.x, i0 = p * TM, t = threadIdx.x; // 256 threads
  __shared__ float wrs[Nn * Rr];        // 16 KB : wr[b,:,:]
  __shared__ float ps[Nn];              //  4 KB : prec[b,:]
  __shared__ float wsv[Nn];             //  4 KB : ww_n[b,:]
  __shared__ float wavepart[TM * 8 * 4];//  8 KB : per-wave fwd partials
  for (int i = t; i < Nn * Rr; i += 256) wrs[i] = wr[(size_t)b * Nn * Rr + i];
  for (int i = t; i < Nn; i += 256){ ps[i] = prec[b * Nn + i]; wsv[i] = wwn[b * Nn + i]; }
  __syncthreads();
  const int jb = t * 4;                 // this thread owns columns jb..jb+3
  float wj[4][4], wwj[4], pj[4];
  #pragma unroll
  for (int e = 0; e < 4; ++e){
    wwj[e] = wsv[jb + e]; pj[e] = ps[jb + e];
    #pragma unroll
    for (int r = 0; r < 4; ++r) wj[e][r] = wrs[(jb + e) * 4 + r];
  }
  float bacc[4][4] = {};
  const int lane = t & 31, wv = t >> 5;
  for (int ii = 0; ii < TM; ++ii){
    int i = i0 + ii;
    float wwi = wsv[i];
    float wri[4];
    #pragma unroll
    for (int r = 0; r < 4; ++r) wri[r] = wrs[i * 4 + r];
    const f4* lrow4 = (const f4*)(link + ((size_t)b * Nn + i) * Nn);
    f4*       Lrow4 = (f4*)(Lout + ((size_t)b * Nn + i) * Nn);
    if (ii + 1 < TM) __builtin_prefetch((const float*)lrow4 + Nn + jb, 0, 1);
    f4 lv4 = __builtin_nontemporal_load(lrow4 + t);   // read-once: NT
    f4 ov;
    float f[4] = {0, 0, 0, 0};
    #pragma unroll
    for (int e = 0; e < 4; ++e){
      int j = jb + e;
      float v = (j == i) ? 0.f : ((1.f - wwi - wwj[e]) * lv4[e] + wwi * pj[e]);
      ov[e] = v;
      #pragma unroll
      for (int r = 0; r < 4; ++r){ f[r] += v * wj[e][r]; bacc[e][r] += v * wri[r]; }
    }
    __builtin_nontemporal_store(ov, Lrow4 + t);       // write-once: NT
    #pragma unroll
    for (int o = 16; o > 0; o >>= 1){
      #pragma unroll
      for (int r = 0; r < 4; ++r) f[r] += __shfl_xor(f[r], o, 32);
    }
    if (lane == 0){
      #pragma unroll
      for (int r = 0; r < 4; ++r) wavepart[(ii * 8 + wv) * 4 + r] = f[r];
    }
  }
  __syncthreads();
  // finish fwd: 64 rows x 4 r, each sums 8 wave partials
  {
    int ii = t >> 2, r = t & 3;     // t in [0,256) covers all (ii, r)
    float sum = 0.f;
    #pragma unroll
    for (int w8 = 0; w8 < 8; ++w8) sum += wavepart[(ii * 8 + w8) * 4 + r];
    fwd[((size_t)b * Nn + i0 + ii) * Rr + r] = sum;
  }
  // deterministic per-block bwd partials straight from registers
  f4* bp4 = (f4*)(bwdpart + ((size_t)(b * 16 + p)) * Nn * Rr);
  #pragma unroll
  for (int e = 0; e < 4; ++e){
    f4 v;
    #pragma unroll
    for (int r = 0; r < 4; ++r) v[r] = bacc[e][r];
    __builtin_nontemporal_store(v, bp4 + t * 4 + e);
  }
}

__global__ void k_bwdred(const float* __restrict__ part, float* __restrict__ bwd){
  int idx = blockIdx.x * blockDim.x + threadIdx.x;  // B*N (f4 granules)
  if (idx >= Bb * Nn) return;
  int b = idx >> 10, q = idx & 1023;
  const f4* p4 = (const f4*)part;
  f4 s = {};
  for (int p = 0; p < 16; ++p) s += p4[((size_t)(b * 16 + p)) * (Nn * Rr / 4) + q];
  ((f4*)bwd)[idx] = s;
}

// ---------------- combine read weights (vectorized over r) ----------------
__global__ void k_wrn(const float* __restrict__ modes, const float* __restrict__ bwd,
                      const float* __restrict__ cr, const float* __restrict__ fwd,
                      float* __restrict__ wrn){
  int idx = blockIdx.x * blockDim.x + threadIdx.x;  // B*N
  if (idx >= Bb * Nn) return;
  int b = idx >> 10;
  f4 pb = *(const f4*)(modes + b * 12);
  f4 pc = *(const f4*)(modes + b * 12 + 4);
  f4 pf = *(const f4*)(modes + b * 12 + 8);
  f4 bw = ((const f4*)bwd)[idx];
  f4 c  = ((const f4*)cr)[idx];
  f4 fw = ((const f4*)fwd)[idx];
  ((f4*)wrn)[idx] = pb * bw + pc * c + pf * fw;
}

// ---------------- read vectors: rv_n[b,w,r] = sum_n Mn[b,n,w]*wrn[b,n,r] ----------------
__global__ void k_rvn(const float* __restrict__ Mn, const float* __restrict__ wrn,
                      float* __restrict__ rvn, float* __restrict__ rfn){
  int b = blockIdx.x; int t = threadIdx.x;  // 256 = 64*4
  int w = t >> 2, r = t & 3;
  float acc = 0.f;
  for (int n = 0; n < Nn; ++n)
    acc += Mn[((size_t)b * Nn + n) * Wd + w] * wrn[((size_t)b * Nn + n) * Rr + r];
  rvn[b * (Wd * Rr) + w * Rr + r] = acc;       // [B, Wd, R]
  rfn[b * (Rr * Wd) + r * Wd + w] = acc;       // flattened [B, R*Wd]
}

// ---------------- WMMA GEMM 3: y = vu + rfn @ W_read ----------------
__global__ void k_y(const float* __restrict__ rfn, const float* __restrict__ Wr,
                    const float* __restrict__ vu, float* __restrict__ y){
  const int j0 = blockIdx.x * 16; const int b0 = blockIdx.y * 16;
  const int l = threadIdx.x;
  const int arow = b0 + (l & 15);
  const int akhi = (l >> 4) * 8;
  const int bcol = j0 + (l & 15);
  const int bko  = (l >> 4) * 8;
  v8f acc = {};
  const float* arr = rfn + (size_t)arow * 256;
  for (int k0 = 0; k0 < 256; k0 += 32){
    v16bf a, bm;
    #pragma unroll
    for (int e = 0; e < 8; ++e){
      a[e]     = (__bf16)arr[k0 + akhi + e];
      a[8 + e] = (__bf16)arr[k0 + 16 + akhi + e];
    }
    #pragma unroll
    for (int e = 0; e < 16; ++e) bm[e] = (__bf16)Wr[(size_t)(k0 + bko + e + (e >= 8 ? 8 : 0)) * Oo + bcol];
    acc = __builtin_amdgcn_wmma_f32_16x16x32_bf16(false, a, false, bm, (short)0, acc, false, false);
  }
  const int ccol = j0 + (l & 15); const int rbase = b0 + (l >> 4) * 8;
  #pragma unroll
  for (int i = 0; i < 8; ++i){
    int row = rbase + i;
    y[row * Oo + ccol] = vu[row * Oo + ccol] + acc[i];
  }
}

// ---------------- host launch ----------------
extern "C" void kernel_launch(void* const* d_in, const int* in_sizes, int n_in,
                              void* d_out, int out_size, void* d_ws, size_t ws_size,
                              hipStream_t stream) {
  const float* x     = (const float*)d_in[0];
  const float* h0    = (const float*)d_in[1];
  const float* c0    = (const float*)d_in[2];
  const float* Wih   = (const float*)d_in[3];
  const float* Whh   = (const float*)d_in[4];
  const float* blstm = (const float*)d_in[5];
  const float* Wv    = (const float*)d_in[6];
  const float* Wxi   = (const float*)d_in[7];
  const float* Wread = (const float*)d_in[8];
  const float* rv    = (const float*)d_in[9];
  const float* M     = (const float*)d_in[10];
  const float* usage = (const float*)d_in[11];
  const float* prec  = (const float*)d_in[12];
  const float* link  = (const float*)d_in[13];
  const float* wr    = (const float*)d_in[14];
  const float* ww    = (const float*)d_in[15];

  float* out = (float*)d_out;
  float* o_y    = out + 0;
  float* o_h    = out + 4096;
  float* o_c    = out + 36864;
  float* o_rvn  = out + 69632;
  float* o_Mn   = out + 86016;
  float* o_u    = out + 4280320;
  float* o_prn  = out + 4345856;
  float* o_L    = out + 4411392;
  float* o_wrn  = out + 71520256;
  float* o_wwn  = out + 71782400;

  float* ws = (float*)d_ws;
  float* w_gates = ws + 0;        // 131072
  float* w_vu    = ws + 131072;   // 4096
  float* w_xi    = ws + 135168;   // 30144
  float* w_rstr  = ws + 165312;   // 256
  float* w_wstr  = ws + 165568;   // 64
  float* w_erase = ws + 165632;   // 4096
  float* w_freeg = ws + 169728;   // 256
  float* w_ag    = ws + 169984;   // 64
  float* w_wg    = ws + 170048;   // 64
  float* w_modes = ws + 170112;   // 768
  float* w_wvec  = ws + 170880;   // 4096
  float* w_a     = ws + 174976;   // 65536
  float* w_cr    = ws + 240512;   // 262144
  float* w_fwd   = ws + 502656;   // 262144
  float* w_bp    = ws + 764800;   // 4194304 (16 partials per batch)
  float* w_bwd   = ws + 4959104;  // 262144
  float* w_rfn   = ws + 5221248;  // 16384

  k_gates<<<dim3(128, 4), 32, 0, stream>>>(x, rv, h0, Wih, Whh, w_gates);
  k_lstm<<<(Bb * Hh + 255) / 256, 256, 0, stream>>>(w_gates, blstm, c0, o_h, o_c);
  k_hproj<<<dim3(34, 4), 32, 0, stream>>>(o_h, Wv, Wxi, w_vu, w_xi);
  k_parse<<<64, 64, 0, stream>>>(w_xi, w_rstr, w_wstr, w_erase, w_freeg, w_ag, w_wg, w_modes, w_wvec);
  k_usage<<<(Bb * Nn + 255) / 256, 256, 0, stream>>>(w_freeg, wr, usage, ww, o_u);
  k_sort_alloc<<<64, 512, 0, stream>>>(o_u, w_a);
  k_write_addr<<<64, 256, 0, stream>>>(M, w_xi, w_a, prec, w_wstr, w_wg, w_ag, o_wwn, o_prn);
  k_mwrite<<<(Bb * Nn * (Wd / 4) + 255) / 256, 256, 0, stream>>>(M, o_wwn, w_erase, w_wvec, o_Mn);
  k_read_addr<<<64, 256, 0, stream>>>(o_Mn, w_xi, w_rstr, w_cr);
  k_link<<<dim3(16, 64), 256, 0, stream>>>(link, prec, o_wwn, wr, o_L, w_fwd, w_bp);
  k_bwdred<<<(Bb * Nn + 255) / 256, 256, 0, stream>>>(w_bp, w_bwd);
  k_wrn<<<(Bb * Nn + 255) / 256, 256, 0, stream>>>(w_modes, w_bwd, w_cr, w_fwd, o_wrn);
  k_rvn<<<64, 256, 0, stream>>>(o_Mn, o_wrn, o_rvn, w_rfn);
  k_y<<<dim3(4, 4), 32, 0, stream>>>(w_rfn, Wread, w_vu, o_y);
}
